// H2Dirichlet_75874892251884
// MI455X (gfx1250) — matrix operand order repaired and checked by reading, not compiled, not run on hardware
//
#include <hip/hip_runtime.h>

typedef __attribute__((ext_vector_type(16))) __bf16 v16bf;
typedef __attribute__((ext_vector_type(8)))  float  v8f;

#define N_ROWS 16384
#define KC     200
#define HD     768
#define BAUG   10

// ---------------------------------------------------------------------------
// Kernel 1: x = hidden @ W^T + b  via v_wmma_f32_16x16x32_bf16
// Block: 256 thr = 8 waves stacked in M -> 128x16 block tile.
// Grid: x = 13 column tiles (13*16 = 208 >= 200), y = 16384/128 = 128.
// Out-of-range B columns read W row 0 (garbage results are never stored;
// WMMA columns are independent, so no contamination of valid columns).
// ---------------------------------------------------------------------------
__global__ __launch_bounds__(256) void gemm_bf16_wmma(
    const float* __restrict__ hidden, const float* __restrict__ W,
    const float* __restrict__ bias, float* __restrict__ x)
{
    const int tid  = threadIdx.x;
    const int wave = tid >> 5;
    const int lane = tid & 31;
    const int half = lane >> 4;   // 0: lanes 0-15, 1: lanes 16-31
    const int l15  = lane & 15;

    const int tileM = blockIdx.y * 128 + wave * 16;
    const int tileN = blockIdx.x * 16;

    const int arow  = tileM + l15;               // A-matrix: row per lane
    const int bcol  = tileN + l15;               // B-matrix: column per lane
    const bool bvalid = (bcol < KC);

    const float* __restrict__ Arow = hidden + (size_t)arow * HD;
    const float* __restrict__ Brow = W + (size_t)(bvalid ? bcol : 0) * HD;

    v8f c = {0.f, 0.f, 0.f, 0.f, 0.f, 0.f, 0.f, 0.f};

    #pragma unroll 4
    for (int h0 = 0; h0 < HD; h0 += 32) {
        v16bf a, bb;
        #pragma unroll
        for (int v = 0; v < 8; ++v) {
            // A (16-bit 16x32): VGPR v<4 -> K = 2v + 8*half; v>=4 -> K = 16 + 2(v-4) + 8*half
            const int ka = ((v < 4) ? (2 * v) : (16 + 2 * (v - 4))) + half * 8;
            const float2 av = *(const float2*)(Arow + h0 + ka);
            a[2 * v]     = (__bf16)av.x;
            a[2 * v + 1] = (__bf16)av.y;
            // B (16-bit 32x16): lane = column, VGPR v<4 -> K = 2v + 8*half; v>=4 -> K = 16 + 2(v-4) + 8*half
            const int kb = ((v < 4) ? (2 * v) : (16 + 2 * (v - 4))) + half * 8;
            const float2 bv = *(const float2*)(Brow + h0 + kb);
            bb[2 * v]     = (__bf16)bv.x;
            bb[2 * v + 1] = (__bf16)bv.y;
        }
        // (neg_a, A, neg_b, B, c_mod, C, reuse_a, reuse_b)
        c = __builtin_amdgcn_wmma_f32_16x16x32_bf16(false, a, false, bb,
                                                    (short)0, c, false, false);
    }

    // C/D layout: VGPR v -> row = tileM + 8*half + v, col = tileN + l15
    if (bvalid) {
        const float bv = bias[bcol];
        #pragma unroll
        for (int v = 0; v < 8; ++v) {
            const int row = tileM + 8 * half + v;
            x[(size_t)row * KC + bcol] = c[v] + bv;
        }
    }
}

// ---------------------------------------------------------------------------
// Kernel 2: per-column mean and 1/sqrt(var+eps) over N rows (x stays in L2:
// the whole 13 MB intermediate fits in the 192 MB global L2).
// Grid: 200 blocks, 256 threads.
// ---------------------------------------------------------------------------
__global__ __launch_bounds__(256) void col_stats(
    const float* __restrict__ x, float* __restrict__ stats)
{
    const int col = blockIdx.x;
    float s = 0.f, s2 = 0.f;
    for (int r = threadIdx.x; r < N_ROWS; r += 256) {
        const float v = x[(size_t)r * KC + col];
        s  += v;
        s2 += v * v;
    }
    __shared__ float sh[256], sh2[256];
    sh[threadIdx.x]  = s;
    sh2[threadIdx.x] = s2;
    __syncthreads();
    for (int off = 128; off > 0; off >>= 1) {
        if (threadIdx.x < off) {
            sh[threadIdx.x]  += sh[threadIdx.x + off];
            sh2[threadIdx.x] += sh2[threadIdx.x + off];
        }
        __syncthreads();
    }
    if (threadIdx.x == 0) {
        const float mu  = sh[0] * (1.f / N_ROWS);
        const float var = sh2[0] * (1.f / N_ROWS) - mu * mu;
        stats[col]      = mu;
        stats[KC + col] = rsqrtf(var + 1e-5f);
    }
}

// ---------------------------------------------------------------------------
// Kernel 3: BN-apply + softplus + Gamma(alpha+10) sample (Marsaglia-Tsang) +
// boosted-uniform product + row-wise Dirichlet normalize.
// Grid: 16384 blocks (one per row), 256 threads (k = col, 200 active).
// Streaming u (131 MB) dominates total HBM traffic -> memory-bound, ~6 us.
// ---------------------------------------------------------------------------
__device__ __forceinline__ unsigned pcg_hash(unsigned v)
{
    v = v * 747796405u + 2891336453u;
    unsigned w = ((v >> ((v >> 28u) + 4u)) ^ v) * 277803737u;
    return (w >> 22u) ^ w;
}

__global__ __launch_bounds__(256) void sample_normalize(
    const float* __restrict__ x, const float* __restrict__ stats,
    const float* __restrict__ u, float* __restrict__ z)
{
    const int n = blockIdx.x;
    const int k = threadIdx.x;
    __shared__ float sh[256];

    // prefetch the 10 strided u planes for this row (global_prefetch_b8)
    if (k < BAUG)
        __builtin_prefetch(u + ((size_t)k * N_ROWS + n) * KC, 0, 1);

    float g = 0.f;
    if (k < KC) {
        const float xn = (x[(size_t)n * KC + k] - stats[k]) * stats[KC + k];
        const float sp = (xn > 20.f) ? xn : log1pf(__expf(xn));
        const float alpha = fmaxf(1e-5f, sp);
        const float aB = alpha + (float)BAUG;

        // Marsaglia-Tsang: a >= 10 => acceptance ~0.99, bounded retries
        const float d  = aB - (1.f / 3.f);
        const float cc = rsqrtf(9.f * d);
        const unsigned seed = ((unsigned)(n * KC + k)) * 2654435761u + 0x9E3779B9u;
        float gam = d;   // fallback near the mode
        #pragma unroll 1
        for (int t = 0; t < 8; ++t) {
            const unsigned h1 = pcg_hash(seed + 3u * (unsigned)t);
            const unsigned h2 = pcg_hash(seed + 3u * (unsigned)t + 1u);
            const unsigned h3 = pcg_hash(seed + 3u * (unsigned)t + 2u);
            const float u1 = (float)(h1 >> 8) * (1.f / 16777216.f) + 5.96e-8f;
            const float u2 = (float)(h2 >> 8) * (1.f / 16777216.f);
            const float r  = __fsqrt_rn(-2.f * __logf(u1));
            const float zn = r * __cosf(6.2831853f * u2);
            float v = 1.f + cc * zn;
            if (v <= 0.f) continue;
            v = v * v * v;
            const float uu = (float)(h3 >> 8) * (1.f / 16777216.f) + 5.96e-8f;
            if (__logf(uu) < 0.5f * zn * zn + d - d * v + d * __logf(v)) {
                gam = d * v;
                break;
            }
        }

        // prod over boosts: u^(1/(alpha+r)) + 1e-10
        float prod = 1.f;
        #pragma unroll
        for (int r2 = 0; r2 < BAUG; ++r2) {
            const float uin = u[((size_t)r2 * N_ROWS + n) * KC + k];
            const float p = __expf(__logf(uin) / (alpha + (float)r2)) + 1e-10f;
            prod *= p;
        }
        g = gam * prod;
    }

    sh[k] = g;
    __syncthreads();
    for (int off = 128; off > 0; off >>= 1) {
        if (k < off) sh[k] += sh[k + off];
        __syncthreads();
    }
    const float total = sh[0];
    if (k < KC) z[(size_t)n * KC + k] = g / total;
}

// ---------------------------------------------------------------------------
extern "C" void kernel_launch(void* const* d_in, const int* in_sizes, int n_in,
                              void* d_out, int out_size, void* d_ws, size_t ws_size,
                              hipStream_t stream)
{
    const float* hidden = (const float*)d_in[0];   // [16384, 768]
    const float* W      = (const float*)d_in[1];   // [200, 768]
    const float* b      = (const float*)d_in[2];   // [200]
    const float* u      = (const float*)d_in[3];   // [10, 16384, 200]
    float* z = (float*)d_out;                      // [16384, 200]

    float* xbuf  = (float*)d_ws;                   // [16384, 200]
    float* stats = xbuf + (size_t)N_ROWS * KC;     // [2, 200] (mu, rsqrt(var+eps))

    dim3 g1(13, N_ROWS / 128);                     // 13 x 128 blocks, 128x16 tiles
    gemm_bf16_wmma<<<g1, 256, 0, stream>>>(hidden, W, b, xbuf);

    col_stats<<<KC, 256, 0, stream>>>(xbuf, stats);

    sample_normalize<<<N_ROWS, 256, 0, stream>>>(xbuf, stats, u, z);
}
